// HGT_17592186044973
// MI455X (gfx1250) — matrix operand order repaired and checked by reading, not compiled, not run on hardware
//
#include <hip/hip_runtime.h>
#include <hip/hip_bf16.h>
#include <math.h>

// ---------------- problem constants (from the reference) ----------------
#define NN   100000
#define RR   4
#define EE   80000
#define IN_  128
#define DH_  256
#define HH   8
#define DK_  32
#define OUT_ 64

typedef _Float16 v8h  __attribute__((ext_vector_type(8)));
typedef _Float16 v16h __attribute__((ext_vector_type(16)));
typedef float    v8f  __attribute__((ext_vector_type(8)));
typedef int      v4i  __attribute__((ext_vector_type(4)));

// ---------------- CDNA5 async global->LDS path (guarded) ----------------
#if __has_builtin(__builtin_amdgcn_global_load_async_to_lds_b128)
#define HAVE_ASYNC_LDS 1
#else
#define HAVE_ASYNC_LDS 0
#endif

__device__ __forceinline__ void async_copy16(const _Float16* g, _Float16* l) {
#if HAVE_ASYNC_LDS
    __builtin_amdgcn_global_load_async_to_lds_b128(
        (v4i*)const_cast<_Float16*>(g), (v4i*)l, 0, 0);
#else
    *(v8h*)l = *(const v8h*)g;   // fallback: global load + ds store
#endif
}

__device__ __forceinline__ void async_wait_all() {
#if HAVE_ASYNC_LDS
#if __has_builtin(__builtin_amdgcn_s_wait_asynccnt)
    __builtin_amdgcn_s_wait_asynccnt(0);
#else
    asm volatile("s_wait_asynccnt 0x0" ::: "memory");
#endif
#endif
}

// ---------------- small helpers ----------------
__device__ __forceinline__ unsigned floatKey(float f) {
    int i = __float_as_int(f);
    unsigned u = (unsigned)i;
    return (i < 0) ? ~u : (u | 0x80000000u);
}
__device__ __forceinline__ float keyFloat(unsigned u) {
    int i = (u & 0x80000000u) ? (int)(u & 0x7FFFFFFFu) : (int)~u;
    return __int_as_float(i);
}
__device__ __forceinline__ float gelu_exact(float x) {
    return 0.5f * x * (1.0f + erff(x * 0.70710678118654752f));
}

// ---------------- utility kernels ----------------
__global__ void zero_f32(float* __restrict__ p, long n) {
    long i = (long)blockIdx.x * blockDim.x + threadIdx.x;
    if (i < n) p[i] = 0.0f;
}

__global__ void cvt_f16(const float* __restrict__ s, _Float16* __restrict__ d,
                        long n, float scale) {
    long i = (long)blockIdx.x * blockDim.x + threadIdx.x;
    if (i < n) d[i] = (_Float16)(s[i] * scale);
}

// Pack a row-major f32 weight [K,NOUT] into WMMA B-fragment order:
// dst[((nt*KT + kt)*32 + lane)*16 + j] = W[kt*32 + (lane>=16?16:0) + j][nt*16 + (lane&15)]
__global__ void pack_weight(const float* __restrict__ src, _Float16* __restrict__ dst,
                            int K, int NOUT) {
    int KT = K >> 5;
    long total = (long)K * NOUT;
    long idx = (long)blockIdx.x * blockDim.x + threadIdx.x;
    if (idx >= total) return;
    int j    = idx & 15;
    int lane = (idx >> 4) & 31;
    long t   = idx >> 9;
    int kt   = (int)(t % KT);
    int nt   = (int)(t / KT);
    int k = kt * 32 + ((lane & 16) ? 16 : 0) + j;
    int n = nt * 16 + (lane & 15);
    dst[idx] = (_Float16)src[(size_t)k * NOUT + n];
}

// ---------------- WMMA GEMM with LDS-staged B panel ----------------
// Block: 256 threads = 8 waves; each wave computes 16 rows x 64 cols per strip.
// B strips (K x 64, contiguous in packed layout) staged global->LDS via async
// b128 copies, then read back as ds_load_b128 fragments.
// MODE 0: plain f32 store to O32
// MODE 1: GELU -> O32 (f32) and O16 (f16)
// MODE 2: mix  -> O32 = val*sigmoid(skip) + Hres*(1-sigmoid(skip))
template <int K, int NOUT, int MODE>
__global__ __launch_bounds__(256)
void gemm_wmma(const _Float16* __restrict__ A, const _Float16* __restrict__ Bp,
               const float* __restrict__ bias,
               float* __restrict__ O32, _Float16* __restrict__ O16,
               const float* __restrict__ Hres, const float* __restrict__ skip,
               int M) {
    constexpr int KT = K / 32;       // k-tiles
    constexpr int NS = NOUT / 64;    // 64-column strips
    __shared__ __align__(16) _Float16 Bsh[K * 64];

    const int tid  = threadIdx.x;
    const int lane = tid & 31;
    const int w    = tid >> 5;
    const int m0   = blockIdx.x * 128 + w * 16;

    int arow = m0 + (lane & 15);
    if (arow >= M) arow = M - 1;          // clamp loads; stores guarded
    const int kofs  = (lane < 16) ? 0 : 16;
    const int rbase = m0 + ((lane < 16) ? 0 : 8);

    float asig = 0.0f;
    if (MODE == 2) asig = 1.0f / (1.0f + expf(-skip[0]));

    for (int s = 0; s < NS; ++s) {
        if (s) __syncthreads();           // previous strip fully consumed

        // stage strip s of packed B (contiguous K*64 halfs) into LDS
        const _Float16* gsrc = Bp + (size_t)s * (4 * KT * 512);
#pragma unroll
        for (int i = 0; i < KT; ++i) {
            int idx = i * 256 + tid;      // 16-byte chunk id
            async_copy16(gsrc + (size_t)idx * 8, Bsh + (size_t)idx * 8);
        }
        async_wait_all();
        __syncthreads();

        v8f c[4] = {};
        for (int kt = 0; kt < KT; ++kt) {
            const _Float16* ap = A + (size_t)arow * K + kt * 32 + kofs;
            v8h alo = *(const v8h*)(ap);
            v8h ahi = *(const v8h*)(ap + 8);
            v16h a;
#pragma unroll
            for (int i = 0; i < 8; ++i) { a[i] = alo[i]; a[8 + i] = ahi[i]; }

#pragma unroll
            for (int nf = 0; nf < 4; ++nf) {
                const _Float16* bp2 = Bsh + ((size_t)(nf * KT + kt) * 32 + lane) * 16;
                v8h blo = *(const v8h*)(bp2);
                v8h bhi = *(const v8h*)(bp2 + 8);
                v16h b;
#pragma unroll
                for (int i = 0; i < 8; ++i) { b[i] = blo[i]; b[8 + i] = bhi[i]; }
                c[nf] = __builtin_amdgcn_wmma_f32_16x16x32_f16(
                    false, a, false, b, (short)0, c[nf], false, false);
            }
        }

        // epilogue for this strip
#pragma unroll
        for (int nf = 0; nf < 4; ++nf) {
            const int col = s * 64 + nf * 16 + (lane & 15);
            const float bc = bias[col];
#pragma unroll
            for (int i = 0; i < 8; ++i) {
                const int r = rbase + i;
                if (r >= M) continue;
                float val = c[nf][i] + bc;
                if (MODE == 0) {
                    O32[(size_t)r * NOUT + col] = val;
                } else if (MODE == 1) {
                    float g = gelu_exact(val);
                    O32[(size_t)r * NOUT + col] = g;
                    O16[(size_t)r * NOUT + col] = (_Float16)g;
                } else { // MODE == 2
                    float o = val * asig + Hres[(size_t)r * NOUT + col] * (1.0f - asig);
                    O32[(size_t)r * NOUT + col] = o;
                }
            }
        }
    }
}

// ---------------- edge phase ----------------
// idx = (r*H + hh)*E + e
__global__ void edge_passA(const float* __restrict__ k, const float* __restrict__ q,
                           const int* __restrict__ src, const int* __restrict__ dst,
                           const float* __restrict__ rel_att,
                           const float* __restrict__ rel_pri,
                           float* __restrict__ score, unsigned* __restrict__ smax) {
    long idx = (long)blockIdx.x * blockDim.x + threadIdx.x;
    if (idx >= (long)RR * EE * HH) return;
    int e  = (int)(idx % EE);
    int hh = (int)((idx / EE) % HH);
    int r  = (int)(idx / ((long)EE * HH));
    int s = src[(size_t)r * EE + e];
    int d = dst[(size_t)r * EE + e];
    const float* kr = k + (size_t)s * DH_ + hh * DK_;
    const float* qr = q + (size_t)d * DH_ + hh * DK_;
    const float* Ar = rel_att + ((size_t)(r * HH + hh) * DK_) * DK_;

    float acc[DK_];
#pragma unroll
    for (int f = 0; f < DK_; ++f) acc[f] = 0.0f;
    for (int dd = 0; dd < DK_; ++dd) {
        float kd = kr[dd];
        const float* row = Ar + dd * DK_;
#pragma unroll
        for (int f = 0; f < DK_; ++f) acc[f] += kd * row[f];
    }
    float sc = 0.0f;
#pragma unroll
    for (int f = 0; f < DK_; ++f) sc += acc[f] * qr[f];
    sc *= rel_pri[r * HH + hh] * 0.17677669529663687f; // 1/sqrt(32)

    score[idx] = sc;
    atomicMax(&smax[((size_t)r * NN + d) * HH + hh], floatKey(sc));
}

__global__ void edge_passB(const int* __restrict__ dst,
                           float* __restrict__ score,
                           const unsigned* __restrict__ smax,
                           float* __restrict__ ssum) {
    long idx = (long)blockIdx.x * blockDim.x + threadIdx.x;
    if (idx >= (long)RR * EE * HH) return;
    int e  = (int)(idx % EE);
    int hh = (int)((idx / EE) % HH);
    int r  = (int)(idx / ((long)EE * HH));
    int d = dst[(size_t)r * EE + e];
    size_t seg = ((size_t)r * NN + d) * HH + hh;
    float ex = expf(score[idx] - keyFloat(smax[seg]));
    score[idx] = ex;
    atomicAdd(&ssum[seg], ex);
}

__global__ void edge_passC(const float* __restrict__ v,
                           const int* __restrict__ src, const int* __restrict__ dst,
                           const float* __restrict__ rel_msg,
                           const float* __restrict__ score,   // holds ex
                           const float* __restrict__ ssum,
                           float* __restrict__ t) {
    long idx = (long)blockIdx.x * blockDim.x + threadIdx.x;
    if (idx >= (long)RR * EE * HH) return;
    int e  = (int)(idx % EE);
    int hh = (int)((idx / EE) % HH);
    int r  = (int)(idx / ((long)EE * HH));
    int s = src[(size_t)r * EE + e];
    int d = dst[(size_t)r * EE + e];
    size_t seg = ((size_t)r * NN + d) * HH + hh;
    float alpha = score[idx] / ssum[seg];

    const float* vr = v + (size_t)s * DH_ + hh * DK_;
    const float* Mr = rel_msg + ((size_t)(r * HH + hh) * DK_) * DK_;
    float acc[DK_];
#pragma unroll
    for (int f = 0; f < DK_; ++f) acc[f] = 0.0f;
    for (int dd = 0; dd < DK_; ++dd) {
        float vd = vr[dd];
        const float* row = Mr + dd * DK_;
#pragma unroll
        for (int f = 0; f < DK_; ++f) acc[f] += vd * row[f];
    }
    float* tr = t + (size_t)d * DH_ + hh * DK_;
#pragma unroll
    for (int f = 0; f < DK_; ++f) atomicAdd(&tr[f], acc[f] * alpha);
}

// ---------------- LayerNorm (wave32 per row) ----------------
__global__ __launch_bounds__(256)
void ln_kernel(const float* __restrict__ x, const float* __restrict__ g,
               const float* __restrict__ b, _Float16* __restrict__ y, int M) {
    int lane = threadIdx.x & 31;
    int wv   = threadIdx.x >> 5;
    int row  = blockIdx.x * 8 + wv;
    if (row >= M) return;
    const float* xr = x + (size_t)row * DH_;
    float v[8];
    float s = 0.0f;
#pragma unroll
    for (int i = 0; i < 8; ++i) { v[i] = xr[lane + i * 32]; s += v[i]; }
#pragma unroll
    for (int o = 16; o; o >>= 1) s += __shfl_xor(s, o, 32);
    float mu = s * (1.0f / DH_);
    float vs = 0.0f;
#pragma unroll
    for (int i = 0; i < 8; ++i) { float dd = v[i] - mu; vs += dd * dd; }
#pragma unroll
    for (int o = 16; o; o >>= 1) vs += __shfl_xor(vs, o, 32);
    float inv = rsqrtf(vs * (1.0f / DH_) + 1e-5f);
    _Float16* yr = y + (size_t)row * DH_;
#pragma unroll
    for (int i = 0; i < 8; ++i) {
        int c = lane + i * 32;
        yr[c] = (_Float16)((v[i] - mu) * inv * g[c] + b[c]);
    }
}

// ---------------- host side ----------------
extern "C" void kernel_launch(void* const* d_in, const int* in_sizes, int n_in,
                              void* d_out, int out_size, void* d_ws, size_t ws_size,
                              hipStream_t stream) {
    const float* feat    = (const float*)d_in[0];
    const int*   src     = (const int*)  d_in[1];
    const int*   dst     = (const int*)  d_in[2];
    const float* W_adapt = (const float*)d_in[3];
    const float* b_adapt = (const float*)d_in[4];
    const float* Wk      = (const float*)d_in[5];
    const float* bk      = (const float*)d_in[6];
    const float* Wv      = (const float*)d_in[7];
    const float* bv      = (const float*)d_in[8];
    const float* Wq      = (const float*)d_in[9];
    const float* bq      = (const float*)d_in[10];
    const float* Wa      = (const float*)d_in[11];
    const float* ba      = (const float*)d_in[12];
    const float* rel_att = (const float*)d_in[13];
    const float* rel_msg = (const float*)d_in[14];
    const float* rel_pri = (const float*)d_in[15];
    const float* skip    = (const float*)d_in[16];
    const float* ln_g    = (const float*)d_in[17];
    const float* ln_b    = (const float*)d_in[18];
    const float* W_out   = (const float*)d_in[19];
    const float* b_out   = (const float*)d_in[20];
    float* out = (float*)d_out;

    // workspace carve (256B aligned)
    size_t off = 0;
    char* base = (char*)d_ws;
    auto take = [&](size_t bytes) -> void* {
        void* p = base + off;
        off = (off + bytes + 255) & ~(size_t)255;
        return p;
    };
    float*     h32    = (float*)    take((size_t)NN * DH_ * 4);
    _Float16*  h16    = (_Float16*) take((size_t)NN * DH_ * 2); // reused as LN output
    float*     k32    = (float*)    take((size_t)NN * DH_ * 4);
    float*     v32    = (float*)    take((size_t)NN * DH_ * 4);
    float*     q32    = (float*)    take((size_t)NN * DH_ * 4); // reused as pre-LN mix
    float*     t32    = (float*)    take((size_t)NN * DH_ * 4);
    _Float16*  t16    = (_Float16*) take((size_t)NN * DH_ * 2);
    _Float16*  feat16 = (_Float16*) take((size_t)NN * IN_ * 2);
    float*     score  = (float*)    take((size_t)RR * EE * HH * 4);
    unsigned*  smax   = (unsigned*) take((size_t)RR * NN * HH * 4);
    float*     ssum   = (float*)    take((size_t)RR * NN * HH * 4);
    _Float16*  pWadapt = (_Float16*)take((size_t)IN_ * DH_ * 2);
    _Float16*  pWk     = (_Float16*)take((size_t)DH_ * DH_ * 2);
    _Float16*  pWv     = (_Float16*)take((size_t)DH_ * DH_ * 2);
    _Float16*  pWq     = (_Float16*)take((size_t)DH_ * DH_ * 2);
    _Float16*  pWa     = (_Float16*)take((size_t)DH_ * DH_ * 2);
    _Float16*  pWout   = (_Float16*)take((size_t)DH_ * OUT_ * 2);
    (void)ws_size; (void)in_sizes; (void)n_in; (void)out_size;

    const int T = 256;
    auto blks = [](long n, int t) { return (int)((n + t - 1) / t); };

    // init accumulators
    zero_f32<<<blks((long)NN * DH_, T), T, 0, stream>>>(t32, (long)NN * DH_);
    zero_f32<<<blks((long)RR * NN * HH, T), T, 0, stream>>>(ssum, (long)RR * NN * HH);
    zero_f32<<<blks((long)RR * NN * HH, T), T, 0, stream>>>((float*)smax, (long)RR * NN * HH);

    // precision staging
    cvt_f16<<<blks((long)NN * IN_, T), T, 0, stream>>>(feat, feat16, (long)NN * IN_, 1.0f);
    pack_weight<<<blks((long)IN_ * DH_, T), T, 0, stream>>>(W_adapt, pWadapt, IN_, DH_);
    pack_weight<<<blks((long)DH_ * DH_, T), T, 0, stream>>>(Wk, pWk, DH_, DH_);
    pack_weight<<<blks((long)DH_ * DH_, T), T, 0, stream>>>(Wv, pWv, DH_, DH_);
    pack_weight<<<blks((long)DH_ * DH_, T), T, 0, stream>>>(Wq, pWq, DH_, DH_);
    pack_weight<<<blks((long)DH_ * DH_, T), T, 0, stream>>>(Wa, pWa, DH_, DH_);
    pack_weight<<<blks((long)DH_ * OUT_, T), T, 0, stream>>>(W_out, pWout, DH_, OUT_);

    const int GB = (NN + 127) / 128;   // 128 rows per block

    // h = GELU(feat @ W_adapt + b)  [WMMA + async-LDS B panel, fused GELU]
    gemm_wmma<IN_, DH_, 1><<<GB, 256, 0, stream>>>(
        feat16, pWadapt, b_adapt, h32, h16, nullptr, nullptr, NN);

    // k, v, q = h @ W* + b*  [WMMA]
    gemm_wmma<DH_, DH_, 0><<<GB, 256, 0, stream>>>(
        h16, pWk, bk, k32, nullptr, nullptr, nullptr, NN);
    gemm_wmma<DH_, DH_, 0><<<GB, 256, 0, stream>>>(
        h16, pWv, bv, v32, nullptr, nullptr, nullptr, NN);
    gemm_wmma<DH_, DH_, 0><<<GB, 256, 0, stream>>>(
        h16, pWq, bq, q32, nullptr, nullptr, nullptr, NN);

    // edge softmax: score+max, exp+sum, scatter message
    long nedge = (long)RR * EE * HH;
    edge_passA<<<blks(nedge, T), T, 0, stream>>>(k32, q32, src, dst, rel_att, rel_pri, score, smax);
    edge_passB<<<blks(nedge, T), T, 0, stream>>>(dst, score, smax, ssum);
    edge_passC<<<blks(nedge, T), T, 0, stream>>>(v32, src, dst, rel_msg, score, ssum, t32);

    // mean over relations + f16 staging
    cvt_f16<<<blks((long)NN * DH_, T), T, 0, stream>>>(t32, t16, (long)NN * DH_, 1.0f / RR);

    // mix = (t @ Wa + ba)*sigmoid(skip) + h*(1-sigmoid(skip))  -> q32 (reused)
    gemm_wmma<DH_, DH_, 2><<<GB, 256, 0, stream>>>(
        t16, pWa, ba, q32, nullptr, h32, skip, NN);

    // LayerNorm -> h16 (reused as f16 staging for final GEMM)
    ln_kernel<<<(NN + 7) / 8, 256, 0, stream>>>(q32, ln_g, ln_b, h16, NN);

    // out = norm @ W_out + b_out  [WMMA]
    gemm_wmma<DH_, OUT_, 0><<<GB, 256, 0, stream>>>(
        h16, pWout, b_out, out, nullptr, nullptr, nullptr, NN);
}
